// PointNetFeaturePropagation_29798483100270
// MI455X (gfx1250) — matrix operand order repaired and checked by reading, not compiled, not run on hardware
//
#include <hip/hip_runtime.h>

#define B_   16
#define N_   4096
#define S_   1024
#define D1_  256
#define D2_  128
#define CIN_ 384
#define C1_  256

typedef __attribute__((ext_vector_type(16))) __bf16 v16bf;
typedef __attribute__((ext_vector_type(8)))  float  v8f;

__device__ __forceinline__ unsigned short f2bf(float f) {
  unsigned u = __float_as_uint(f);
  u = (u + 0x7FFFu + ((u >> 16) & 1u)) >> 16;   // round-to-nearest-even
  return (unsigned short)u;
}

// ---------------------------------------------------------------- 3-NN interp
__global__ void k_three_interp(const float* __restrict__ xyz1,
                               const float* __restrict__ xyz2,
                               const float* __restrict__ points2,
                               float* __restrict__ interp) {
  __shared__ float sx[S_], sy[S_], sz[S_];
  const int b = blockIdx.y;
  const int tid = threadIdx.x;
  for (int t = tid; t < S_; t += blockDim.x) {
    sx[t] = xyz2[((size_t)b * 3 + 0) * S_ + t];
    sy[t] = xyz2[((size_t)b * 3 + 1) * S_ + t];
    sz[t] = xyz2[((size_t)b * 3 + 2) * S_ + t];
  }
  __syncthreads();

  const int n = blockIdx.x * blockDim.x + tid;
  const float px = xyz1[((size_t)b * 3 + 0) * N_ + n];
  const float py = xyz1[((size_t)b * 3 + 1) * N_ + n];
  const float pz = xyz1[((size_t)b * 3 + 2) * N_ + n];

  float d0 = 3.4e38f, d1 = 3.4e38f, d2 = 3.4e38f;
  int   i0 = 0, i1 = 0, i2 = 0;
  for (int s = 0; s < S_; ++s) {
    const float dx = px - sx[s], dy = py - sy[s], dz = pz - sz[s];
    const float d = dx * dx + dy * dy + dz * dz;      // same ordering as sqrt(d)
    if (d < d0)      { d2 = d1; i2 = i1; d1 = d0; i1 = i0; d0 = d; i0 = s; }
    else if (d < d1) { d2 = d1; i2 = i1; d1 = d;  i1 = s; }
    else if (d < d2) { d2 = d;  i2 = s; }
  }
  float w0 = 1.f / (sqrtf(d0) + 1e-10f);
  float w1 = 1.f / (sqrtf(d1) + 1e-10f);
  float w2 = 1.f / (sqrtf(d2) + 1e-10f);
  const float ws = 1.f / (w0 + w1 + w2);
  w0 *= ws; w1 *= ws; w2 *= ws;

  const float* p2 = points2 + (size_t)b * D2_ * S_;
  float* op = interp + (size_t)b * D2_ * N_ + n;
  for (int d = 0; d < D2_; ++d) {
    const float* row = p2 + (size_t)d * S_;
    op[(size_t)d * N_] = w0 * row[i0] + w1 * row[i1] + w2 * row[i2];
  }
}

// ---------------------------------------------------------------- GEMM 1
// y1[b,m,n] = sum_k W1[m,k] * x[b,k,n] + b1[m];  x = [points1 ; interp]
// block: 512 thr = 16 waves, tile 256M x 64N -> each activation byte read once
__global__ void k_gemm1(const float* __restrict__ points1,
                        const float* __restrict__ interp,
                        const float* __restrict__ W1,
                        const float* __restrict__ bias1,
                        float* __restrict__ y1,
                        float* __restrict__ stats) {
  __shared__ unsigned short lA[256][34];   // [m][k] bf16, +pad (keeps u32 align)
  __shared__ unsigned short lB[64][34];    // [n][k] bf16 (transposed stage)

  const int tid  = threadIdx.x;
  const int lane = tid & 31, w = tid >> 5;
  const int ll = lane & 15, lh = lane >> 4;
  const int b  = blockIdx.z;
  const int n0 = blockIdx.x * 64;

  const v8f vzero = {0.f, 0.f, 0.f, 0.f, 0.f, 0.f, 0.f, 0.f};
  v8f acc[4];
#pragma unroll
  for (int j = 0; j < 4; ++j) acc[j] = vzero;

  for (int kk = 0; kk < CIN_; kk += 32) {
    // uniform per-step source select (kk multiple of 32, boundary at 256)
    const float* src = (kk < D1_)
        ? points1 + ((size_t)b * D1_ + kk) * N_
        : interp  + ((size_t)b * D2_ + (kk - D1_)) * N_;

    for (int t = tid; t < 256 * 32; t += 512) {
      const int m = t >> 5, k = t & 31;
      lA[m][k] = f2bf(W1[(size_t)m * CIN_ + kk + k]);
    }
    for (int t = tid; t < 32 * 64; t += 512) {
      const int k = t >> 6, n = t & 63;
      lB[n][k] = f2bf(src[(size_t)k * N_ + n0 + n]);
    }
    // prefetch next K-step's activation tile into cache (global_prefetch_b8)
    if (kk + 32 < CIN_) {
      const float* nsrc = (kk + 32 < D1_)
          ? points1 + ((size_t)b * D1_ + kk + 32) * N_
          : interp  + ((size_t)b * D2_ + (kk + 32 - D1_)) * N_;
      for (int t = tid; t < 32 * 64; t += 512)
        __builtin_prefetch(&nsrc[(size_t)(t >> 6) * N_ + n0 + (t & 63)], 0, 1);
    }
    __syncthreads();

    // A fragment: 16x32 bf16, ISA layout: lanes 0-15 K={0..7,16..23}, 16-31 +8
    union { v16bf v; unsigned u[8]; } fa;
#pragma unroll
    for (int i = 0; i < 8; ++i) {
      const int k = (i < 4) ? (lh * 8 + 2 * i) : (16 + lh * 8 + 2 * (i - 4));
      fa.u[i] = *(const unsigned*)&lA[16 * w + ll][k];
    }
#pragma unroll
    for (int j = 0; j < 4; ++j) {
      // B fragment: 32x16, lanes 0-15 K={0..7,16..23}, 16-31 +8 (matches A)
      union { v16bf v; unsigned u[8]; } fb;
#pragma unroll
      for (int i = 0; i < 8; ++i)
        fb.u[i] = *(const unsigned*)&lB[16 * j + ll][(i < 4) ? (lh * 8 + 2 * i) : (16 + lh * 8 + 2 * (i - 4))];
      acc[j] = __builtin_amdgcn_wmma_f32_16x16x32_bf16(
          false, fa.v, false, fb.v, (short)0, acc[j], false, false);
    }
    __syncthreads();
  }

  // epilogue: bias, store y1, per-channel sum / sumsq (BN train stats)
  float* sum1 = stats;          // [256]
  float* sq1  = stats + 256;    // [256]
#pragma unroll
  for (int r = 0; r < 8; ++r) {
    const int mg = 16 * w + r + 8 * lh;        // C/D layout: half-wave offset 8
    const float bv = bias1[mg];
    float s = 0.f, s2 = 0.f;
#pragma unroll
    for (int j = 0; j < 4; ++j) {
      const float v = acc[j][r] + bv;
      y1[((size_t)b * C1_ + mg) * N_ + (n0 + 16 * j + ll)] = v;
      s += v; s2 += v * v;
    }
#pragma unroll
    for (int o = 1; o < 16; o <<= 1) {         // reduce 16 lanes sharing mg
      s  += __shfl_xor(s,  o, 32);
      s2 += __shfl_xor(s2, o, 32);
    }
    if (ll == 0) { atomicAdd(&sum1[mg], s); atomicAdd(&sq1[mg], s2); }
  }
}

// ---------------------------------------------------------------- BN finalize
__global__ void k_bn_fin(float* __restrict__ stats,
                         const float* __restrict__ g,
                         const float* __restrict__ beta, int which) {
  const int c = threadIdx.x;
  const float inv = 1.f / 65536.f;             // B*N samples per channel
  const float mean = stats[which * 512 + c] * inv;
  const float var  = stats[which * 512 + 256 + c] * inv - mean * mean;
  const float a    = g[c] * rsqrtf(var + 1e-5f);
  stats[1024 + which * 512 + c]       = a;
  stats[1024 + which * 512 + 256 + c] = beta[c] - a * mean;
}

// ---------------------------------------------------------------- GEMM 2
// out_raw[b,m,n] = sum_k W2[m,k] * relu(a1[k]*y1[b,k,n]+c1[k]) + b2[m]
__global__ void k_gemm2(const float* __restrict__ y1,
                        const float* __restrict__ W2,
                        const float* __restrict__ bias2,
                        float* __restrict__ stats,
                        float* __restrict__ out) {
  __shared__ unsigned short lA[256][34];
  __shared__ unsigned short lB[64][34];

  const int tid  = threadIdx.x;
  const int lane = tid & 31, w = tid >> 5;
  const int ll = lane & 15, lh = lane >> 4;
  const int b  = blockIdx.z;
  const int n0 = blockIdx.x * 64;
  const float* a1 = stats + 1024;
  const float* c1 = stats + 1280;

  const v8f vzero = {0.f, 0.f, 0.f, 0.f, 0.f, 0.f, 0.f, 0.f};
  v8f acc[4];
#pragma unroll
  for (int j = 0; j < 4; ++j) acc[j] = vzero;

  for (int kk = 0; kk < C1_; kk += 32) {
    for (int t = tid; t < 256 * 32; t += 512) {
      const int m = t >> 5, k = t & 31;
      lA[m][k] = f2bf(W2[(size_t)m * C1_ + kk + k]);
    }
    for (int t = tid; t < 32 * 64; t += 512) {
      const int k = t >> 6, n = t & 63;
      const int kg = kk + k;
      const float yv = y1[((size_t)b * C1_ + kg) * N_ + n0 + n];
      lB[n][k] = f2bf(fmaxf(a1[kg] * yv + c1[kg], 0.f));   // fused BN1+ReLU
    }
    if (kk + 32 < C1_) {
      for (int t = tid; t < 32 * 64; t += 512)
        __builtin_prefetch(
            &y1[((size_t)b * C1_ + kk + 32 + (t >> 6)) * N_ + n0 + (t & 63)], 0, 1);
    }
    __syncthreads();

    union { v16bf v; unsigned u[8]; } fa;
#pragma unroll
    for (int i = 0; i < 8; ++i) {
      const int k = (i < 4) ? (lh * 8 + 2 * i) : (16 + lh * 8 + 2 * (i - 4));
      fa.u[i] = *(const unsigned*)&lA[16 * w + ll][k];
    }
#pragma unroll
    for (int j = 0; j < 4; ++j) {
      union { v16bf v; unsigned u[8]; } fb;
#pragma unroll
      for (int i = 0; i < 8; ++i)
        fb.u[i] = *(const unsigned*)&lB[16 * j + ll][(i < 4) ? (lh * 8 + 2 * i) : (16 + lh * 8 + 2 * (i - 4))];
      acc[j] = __builtin_amdgcn_wmma_f32_16x16x32_bf16(
          false, fa.v, false, fb.v, (short)0, acc[j], false, false);
    }
    __syncthreads();
  }

  float* sum2 = stats + 512;
  float* sq2  = stats + 768;
#pragma unroll
  for (int r = 0; r < 8; ++r) {
    const int mg = 16 * w + r + 8 * lh;
    const float bv = bias2[mg];
    float s = 0.f, s2 = 0.f;
#pragma unroll
    for (int j = 0; j < 4; ++j) {
      const float v = acc[j][r] + bv;
      out[((size_t)b * C1_ + mg) * N_ + (n0 + 16 * j + ll)] = v;
      s += v; s2 += v * v;
    }
#pragma unroll
    for (int o = 1; o < 16; o <<= 1) {
      s  += __shfl_xor(s,  o, 32);
      s2 += __shfl_xor(s2, o, 32);
    }
    if (ll == 0) { atomicAdd(&sum2[mg], s); atomicAdd(&sq2[mg], s2); }
  }
}

// ---------------------------------------------------------------- BN2 apply
__global__ void k_apply(float* __restrict__ out, const float* __restrict__ stats) {
  const size_t i = (size_t)blockIdx.x * 256 + threadIdx.x;
  const int c = (int)((i >> 12) & 255);          // [b][c][n], n-stride 4096
  out[i] = fmaxf(stats[1536 + c] * out[i] + stats[1792 + c], 0.f);
}

__global__ void k_zero(float* __restrict__ stats) {
  const int t = threadIdx.x;
#pragma unroll
  for (int i = 0; i < 4; ++i) stats[i * 256 + t] = 0.f;
}

// ---------------------------------------------------------------- launcher
extern "C" void kernel_launch(void* const* d_in, const int* in_sizes, int n_in,
                              void* d_out, int out_size, void* d_ws, size_t ws_size,
                              hipStream_t stream) {
  const float* xyz1    = (const float*)d_in[0];
  const float* xyz2    = (const float*)d_in[1];
  const float* points1 = (const float*)d_in[2];
  const float* points2 = (const float*)d_in[3];
  const float* W1      = (const float*)d_in[4];
  const float* b1      = (const float*)d_in[5];
  const float* g1      = (const float*)d_in[6];
  const float* beta1   = (const float*)d_in[7];
  const float* W2      = (const float*)d_in[8];
  const float* b2      = (const float*)d_in[9];
  const float* g2      = (const float*)d_in[10];
  const float* beta2   = (const float*)d_in[11];

  float* interp = (float*)d_ws;                             // 16*128*4096 f32
  float* y1     = interp + (size_t)B_ * D2_ * N_;           // 16*256*4096 f32
  float* stats  = y1 + (size_t)B_ * C1_ * N_;               // 2048 f32
  float* out    = (float*)d_out;

  k_zero<<<1, 256, 0, stream>>>(stats);
  k_three_interp<<<dim3(N_ / 128, B_), 128, 0, stream>>>(xyz1, xyz2, points2, interp);
  k_gemm1<<<dim3(N_ / 64, 1, B_), 512, 0, stream>>>(points1, interp, W1, b1, y1, stats);
  k_bn_fin<<<1, 256, 0, stream>>>(stats, g1, beta1, 0);
  k_gemm2<<<dim3(N_ / 64, 1, B_), 512, 0, stream>>>(y1, W2, b2, stats, out);
  k_bn_fin<<<1, 256, 0, stream>>>(stats, g2, beta2, 1);
  k_apply<<<(B_ * C1_ * N_) / 256, 256, 0, stream>>>(out, stats);
}
